// Encoder_83021717832656
// MI455X (gfx1250) — matrix operand order repaired and checked by reading, not compiled, not run on hardware
//
#include <hip/hip_runtime.h>
#include <hip/hip_bf16.h>

#define NNODES 50000
#define NEDGES 800000
#define NGRAPHS 64
#define FDIM 128

typedef __attribute__((ext_vector_type(16))) __bf16 v16bf;
typedef __attribute__((ext_vector_type(8)))  __bf16 v8bf;
typedef __attribute__((ext_vector_type(8)))  float  v8f;

union ABfrag { v16bf v; v8bf h[2]; };

// ---------------- utility kernels ----------------

__global__ void zero_kernel(float* __restrict__ p, int n) {
    int i = blockIdx.x * blockDim.x + threadIdx.x;
    int stride = gridDim.x * blockDim.x;
    for (; i < n; i += stride) p[i] = 0.0f;
}

__global__ void degree_kernel(const int* __restrict__ dst, float* __restrict__ deg) {
    int e = blockIdx.x * blockDim.x + threadIdx.x;
    if (e < NEDGES) atomicAdd(&deg[dst[e]], 1.0f);
}

__global__ void norm_finalize(float* __restrict__ nrm) {
    int i = blockIdx.x * blockDim.x + threadIdx.x;
    if (i < NNODES) nrm[i] = rsqrtf(fmaxf(nrm[i], 1.0f));
}

// out[dst] += x[src] * norm[src]  (one wave per edge, float4 per lane)
__global__ void spmm_scatter(const float* __restrict__ x,
                             const int* __restrict__ src,
                             const int* __restrict__ dst,
                             const float* __restrict__ nrm,
                             float* __restrict__ out) {
    unsigned tid = blockIdx.x * blockDim.x + threadIdx.x;
    unsigned e = tid >> 5;        // edge index (uniform within a wave)
    unsigned lane = tid & 31;     // 32 lanes x float4 = 128 features
    if (e >= NEDGES) return;
    int s = src[e];
    int d = dst[e];
    float ns = nrm[s];
    float4 v = ((const float4*)(x + (size_t)s * FDIM))[lane];
    float* o = out + (size_t)d * FDIM + lane * 4;
    atomicAdd(o + 0, v.x * ns);
    atomicAdd(o + 1, v.y * ns);
    atomicAdd(o + 2, v.z * ns);
    atomicAdd(o + 3, v.w * ns);
}

// y[i,f] *= norm[i]
__global__ void row_scale(float* __restrict__ y, const float* __restrict__ nrm) {
    int i = blockIdx.x * blockDim.x + threadIdx.x;
    if (i < NNODES * FDIM) y[i] *= nrm[i >> 7];
}

// ---------------- WMMA GEMM: relu([h0|h1|h2] @ W + b) ----------------
// A tile M=128 per block, K = 384 in 12 steps of 32, N = OUT_DIM (128 or 64).
// 8 waves/block; wave w computes rows 16w..16w+15 x all N tiles.
template<int OUT_DIM, int READOUT>
__global__ __launch_bounds__(256)
void tag_gemm(const float* __restrict__ h0, const float* __restrict__ h1,
              const float* __restrict__ h2, const float* __restrict__ W,
              const float* __restrict__ bias, float* __restrict__ C,
              const int* __restrict__ gids, float* __restrict__ gout) {
    constexpr int NT = OUT_DIM / 16;   // 16-wide N tiles per wave
    constexpr int LDST = 40;           // padded bf16 stride (80B, 16B aligned rows)

    __shared__ __bf16 ldsA[128][LDST];
    __shared__ __bf16 ldsB[OUT_DIM][LDST];

    const int t    = threadIdx.x;
    const int w    = t >> 5;
    const int lane = t & 31;
    const int lr   = lane & 15;   // N column within tile / M row within wave tile
    const int lg   = lane >> 4;   // lane group (0 or 1)
    const int mBase = blockIdx.x * 128;

    v8f acc[NT];
    for (int i = 0; i < NT; ++i) acc[i] = (v8f)0.0f;

    for (int kt = 0; kt < 12; ++kt) {
        const int k0 = kt * 32;
        const float* A = (k0 < 128) ? h0 : (k0 < 256 ? h1 : h2);
        const int c0 = k0 & 127;

        // stage A: 128 rows x 32 cols f32 -> bf16 (1024 float4, 4 per thread)
        for (int i = 0; i < 4; ++i) {
            int idx = t + i * 256;         // 0..1023
            int row = idx >> 3;            // 8 float4 per row
            int cq  = (idx & 7) * 4;
            float4 v = make_float4(0.f, 0.f, 0.f, 0.f);
            int grow = mBase + row;
            if (grow < NNODES)
                v = *(const float4*)(A + (size_t)grow * FDIM + c0 + cq);
            __bf16* p = &ldsA[row][cq];
            p[0] = (__bf16)v.x; p[1] = (__bf16)v.y;
            p[2] = (__bf16)v.z; p[3] = (__bf16)v.w;
        }
        // stage B transposed: ldsB[n][k] = W[(k0+k)*OUT_DIM + n]
        for (int e = t; e < 32 * OUT_DIM; e += 256) {
            int k = e / OUT_DIM;
            int n = e - k * OUT_DIM;
            ldsB[n][k] = (__bf16)W[(size_t)(k0 + k) * OUT_DIM + n];
        }
        __syncthreads();

        // A fragment (16-bit A 16x32 layout): lane lr row, K = {lg*8..+7, 16+lg*8..+7}
        const int arow = (w << 4) + lr;
        ABfrag a;
        a.h[0] = *(const v8bf*)&ldsA[arow][lg * 8];
        a.h[1] = *(const v8bf*)&ldsA[arow][16 + lg * 8];

        for (int nt = 0; nt < NT; ++nt) {
            // B fragment (16-bit B 32x16): lane lr = column N, K = {lg*8..+7, 16+lg*8..+7}
            ABfrag b;
            b.h[0] = *(const v8bf*)&ldsB[nt * 16 + lr][lg * 8];
            b.h[1] = *(const v8bf*)&ldsB[nt * 16 + lr][16 + lg * 8];
            acc[nt] = __builtin_amdgcn_wmma_f32_16x16x32_bf16(
                false, a.v, false, b.v, (short)0, acc[nt], false, false);
        }
        __syncthreads();
    }

    // epilogue: bias + relu, then store or fused segment-max readout
    for (int nt = 0; nt < NT; ++nt) {
        const int col = nt * 16 + lr;
        const float bv = bias[col];
        for (int r = 0; r < 8; ++r) {
            const int row = mBase + (w << 4) + lg * 8 + r;
            if (row < NNODES) {
                float v = fmaxf(acc[nt][r] + bv, 0.0f);
                if (READOUT) {
                    // v >= 0, so float order == uint order
                    int g = gids[row];
                    atomicMax((unsigned int*)&gout[(size_t)g * OUT_DIM + col],
                              __float_as_uint(v));
                } else {
                    C[(size_t)row * OUT_DIM + col] = v;
                }
            }
        }
    }
}

// ---------------- launch ----------------

extern "C" void kernel_launch(void* const* d_in, const int* in_sizes, int n_in,
                              void* d_out, int out_size, void* d_ws, size_t ws_size,
                              hipStream_t stream) {
    (void)in_sizes; (void)n_in; (void)out_size; (void)ws_size;
    const float* h   = (const float*)d_in[0];
    const int*   src = (const int*)d_in[1];
    const int*   dst = (const int*)d_in[2];
    const int*   gid = (const int*)d_in[3];
    const float* W1  = (const float*)d_in[4];
    const float* b1  = (const float*)d_in[5];
    const float* W2  = (const float*)d_in[6];
    const float* b2  = (const float*)d_in[7];
    float* out = (float*)d_out;

    // workspace layout (bytes, 256-aligned)
    char* ws = (char*)d_ws;
    float* nrm  = (float*)(ws);                                 // 50000 f32
    float* bufA = (float*)(ws + 200192);                        // [50000,128]
    float* bufB = (float*)(ws + 200192 + 25600000);             // [50000,128]
    float* bufX = (float*)(ws + 200192 + 2 * 25600000);         // [50000,128]

    const int feat = NNODES * FDIM;
    dim3 blk(256);
    dim3 gZeroN((NNODES + 255) / 256);
    dim3 gZeroF((feat + 255) / 256);
    dim3 gEdge((NEDGES + 255) / 256);
    dim3 gScat(((size_t)NEDGES * 32 + 255) / 256);
    dim3 gGemm((NNODES + 127) / 128);

    // norm = clip(deg,1)^-0.5
    zero_kernel<<<gZeroN, blk, 0, stream>>>(nrm, NNODES);
    degree_kernel<<<gEdge, blk, 0, stream>>>(dst, nrm);
    norm_finalize<<<gZeroN, blk, 0, stream>>>(nrm);

    // conv1 hops: bufA = norm * A(norm*h), bufB = norm * A(norm*bufA)
    zero_kernel<<<gZeroF, blk, 0, stream>>>(bufA, feat);
    spmm_scatter<<<gScat, blk, 0, stream>>>(h, src, dst, nrm, bufA);
    row_scale<<<gZeroF, blk, 0, stream>>>(bufA, nrm);
    zero_kernel<<<gZeroF, blk, 0, stream>>>(bufB, feat);
    spmm_scatter<<<gScat, blk, 0, stream>>>(bufA, src, dst, nrm, bufB);
    row_scale<<<gZeroF, blk, 0, stream>>>(bufB, nrm);

    // conv1 GEMM: bufX = relu([h|bufA|bufB] @ W1 + b1)   [50000,128]
    tag_gemm<128, 0><<<gGemm, blk, 0, stream>>>(h, bufA, bufB, W1, b1,
                                                bufX, nullptr, nullptr);

    // conv2 hops on bufX
    zero_kernel<<<gZeroF, blk, 0, stream>>>(bufA, feat);
    spmm_scatter<<<gScat, blk, 0, stream>>>(bufX, src, dst, nrm, bufA);
    row_scale<<<gZeroF, blk, 0, stream>>>(bufA, nrm);
    zero_kernel<<<gZeroF, blk, 0, stream>>>(bufB, feat);
    spmm_scatter<<<gScat, blk, 0, stream>>>(bufA, src, dst, nrm, bufB);
    row_scale<<<gZeroF, blk, 0, stream>>>(bufB, nrm);

    // readout init (relu output >= 0, so 0.0f is the identity for max)
    dim3 gZeroO((NGRAPHS * 64 + 255) / 256);
    zero_kernel<<<gZeroO, blk, 0, stream>>>(out, NGRAPHS * 64);

    // conv2 GEMM + fused segment-max readout into d_out [64,64]
    tag_gemm<64, 1><<<gGemm, blk, 0, stream>>>(bufX, bufA, bufB, W2, b2,
                                               nullptr, gid, out);
}
